// MambaBlock_30743375905081
// MI455X (gfx1250) — compile-verified, matrix operand order checked by reading, not run on hardware
//
#include <hip/hip_runtime.h>
#include <hip/hip_bf16.h>

// ---- problem constants (from reference) ----
#define BSZ   2
#define LQ    2048
#define DM    1024
#define DI    2048          // d_inner
#define NST   16            // d_state
#define DTR   64            // dt_rank
#define PRJ   (DTR + 2*NST) // 96
#define ROWS  (BSZ*LQ)      // 4096

typedef __attribute__((ext_vector_type(16))) __bf16 v16bf;
typedef __attribute__((ext_vector_type(8)))  __bf16 v8bf;
typedef __attribute__((ext_vector_type(8)))  float  v8f;

__device__ __forceinline__ void bf16_split(float x, __bf16& hi, __bf16& lo) {
    __bf16 h = (__bf16)x;
    hi = h;
    lo = (__bf16)(x - (float)h);
}

// ---------------------------------------------------------------------------
// Transpose + bf16-split a weight matrix W[K x N] (row-major, fp32) into
// Wt_hi / Wt_lo [N x K] (row-major, bf16). K, N multiples of 32.
// Puts the WMMA B operand in a layout where each lane's fragment is two
// contiguous 16-byte runs -> pure global_load_b128 in the GEMM, zero VALU.
// ---------------------------------------------------------------------------
__global__ __launch_bounds__(256)
void tsplit_kernel(const float* __restrict__ W, __bf16* __restrict__ th,
                   __bf16* __restrict__ tl, int K, int N)
{
    __shared__ float tile[32][33];
    const int k0 = blockIdx.y * 32;
    const int n0 = blockIdx.x * 32;
    const int tx = threadIdx.x & 31;
    const int ty = threadIdx.x >> 5;          // 0..7
#pragma unroll
    for (int r = ty; r < 32; r += 8)
        tile[r][tx] = W[(size_t)(k0 + r) * N + n0 + tx];
    __syncthreads();
#pragma unroll
    for (int r = ty; r < 32; r += 8) {
        float v = tile[tx][r];                // W[k0+tx][n0+r]
        __bf16 h, l;
        bf16_split(v, h, l);
        size_t o = (size_t)(n0 + r) * K + k0 + tx;
        th[o] = h;
        tl[o] = l;
    }
}

// ---------------------------------------------------------------------------
// GEMM C[M,N] = A[M,K] @ B[K,N], A fp32 (split on the fly), B pre-split and
// pre-transposed (Bt[N,K] bf16 hi/lo). fp32-class accuracy via 3-term bf16
// product (hi*hi + hi*lo + lo*hi), fp32 WMMA accumulate.
// One wave computes a (MT*16) x (NT*16) tile: MT*NT accumulators, A fragments
// reused across NT column tiles. Requires M % (MT*16) == 0, N % (NT*16) == 0,
// K % 32 == 0 (true for all calls here).
// ---------------------------------------------------------------------------
template <int MT, int NT>
__global__ __launch_bounds__(32)
void gemm_wmma(const float* __restrict__ A, const __bf16* __restrict__ Bth,
               const __bf16* __restrict__ Btl, float* __restrict__ C,
               int K, int lda, int ldc)
{
    const int lane = threadIdx.x;             // 0..31, all active (EXEC all-1)
    const int lid  = lane & 15;
    const int half = lane >> 4;
    const int kb   = half * 8;                // per-half k sub-base
    const int row0 = blockIdx.y * (MT * 16);
    const int col0 = blockIdx.x * (NT * 16);

    v8f acc[MT][NT] = {};

    for (int k0 = 0; k0 < K; k0 += 32) {
        // ---- A fragments: load fp32, split to bf16 hi/lo ----
        v16bf ah[MT], al[MT];
#pragma unroll
        for (int mt = 0; mt < MT; ++mt) {
            const float* Ar = A + (size_t)(row0 + mt * 16 + lid) * lda + k0 + kb;
            __builtin_prefetch(Ar + 64, 0, 1);     // global_prefetch_b8
#pragma unroll
            for (int j = 0; j < 8; ++j) {
                __bf16 h, l;
                bf16_split(Ar[j],      h, l); ah[mt][j]     = h; al[mt][j]     = l;
                bf16_split(Ar[16 + j], h, l); ah[mt][8 + j] = h; al[mt][8 + j] = l;
            }
        }
        // ---- B fragments: straight b128 loads of pre-split bf16 ----
#pragma unroll
        for (int nt = 0; nt < NT; ++nt) {
            size_t bo = (size_t)(col0 + nt * 16 + lid) * K + k0 + kb;
            v8bf h0 = *(const v8bf*)(Bth + bo);
            v8bf h1 = *(const v8bf*)(Bth + bo + 16);
            v8bf l0 = *(const v8bf*)(Btl + bo);
            v8bf l1 = *(const v8bf*)(Btl + bo + 16);
            v16bf bh = __builtin_shufflevector(h0, h1, 0,1,2,3,4,5,6,7,
                                               8,9,10,11,12,13,14,15);
            v16bf bl = __builtin_shufflevector(l0, l1, 0,1,2,3,4,5,6,7,
                                               8,9,10,11,12,13,14,15);
#pragma unroll
            for (int mt = 0; mt < MT; ++mt) {
                acc[mt][nt] = __builtin_amdgcn_wmma_f32_16x16x32_bf16(
                    false, ah[mt], false, bh, (short)0, acc[mt][nt], false, false);
                acc[mt][nt] = __builtin_amdgcn_wmma_f32_16x16x32_bf16(
                    false, ah[mt], false, bl, (short)0, acc[mt][nt], false, false);
                acc[mt][nt] = __builtin_amdgcn_wmma_f32_16x16x32_bf16(
                    false, al[mt], false, bh, (short)0, acc[mt][nt], false, false);
            }
        }
    }
    // ---- C/D layout: VGPR v -> M = half*8 + v, N = lid ----
#pragma unroll
    for (int mt = 0; mt < MT; ++mt)
#pragma unroll
        for (int nt = 0; nt < NT; ++nt) {
            float* Cr = C + (size_t)(row0 + mt * 16 + half * 8) * ldc
                          + col0 + nt * 16 + lid;
#pragma unroll
            for (int v = 0; v < 8; ++v)
                Cr[(size_t)v * ldc] = acc[mt][nt][v];
        }
}

// ---------------------------------------------------------------------------
// Causal depthwise conv1d (K=4) + SiLU on the x-branch; rewrites the z-branch
// of xz with silu(z) in place (consumed later by the scan).
// ---------------------------------------------------------------------------
__global__ __launch_bounds__(256)
void conv_silu_kernel(float* __restrict__ xz, const float* __restrict__ cw,
                      const float* __restrict__ cb, float* __restrict__ xc)
{
    int i = blockIdx.x * 256 + threadIdx.x;     // over ROWS*DI
    if (i >= ROWS * DI) return;
    int d  = i % DI;
    int bl = i / DI;                            // b*LQ + l
    int l  = bl % LQ;

    float acc = cb[d];
#pragma unroll
    for (int k = 0; k < 4; ++k) {
        int t = l + k - 3;
        if (t >= 0)
            acc += cw[d * 4 + k] * xz[(size_t)(bl + k - 3) * (2 * DI) + d];
    }
    xc[i] = acc / (1.f + __expf(-acc));         // silu

    size_t zi = (size_t)bl * (2 * DI) + DI + d;
    float z = xz[zi];
    xz[zi] = z / (1.f + __expf(-z));            // silu(z) in place
}

// ---------------------------------------------------------------------------
// dt = softplus(dt_lin + b_dt), in place
// ---------------------------------------------------------------------------
__global__ __launch_bounds__(256)
void softplus_kernel(float* __restrict__ dt, const float* __restrict__ b_dt)
{
    int i = blockIdx.x * 256 + threadIdx.x;
    if (i >= ROWS * DI) return;
    float x = dt[i] + b_dt[i % DI];
    dt[i] = (x > 20.f) ? x : log1pf(__expf(x));
}

// ---------------------------------------------------------------------------
// Selective scan: one lane owns one (b,d) channel; h[16] lives in VGPRs.
// B/C (shared by every channel at a timestep) are staged into LDS in chunks.
// y = (scan + u*D) * silu(z), written over xc in place.
// ---------------------------------------------------------------------------
#define SCAN_T 64
__global__ __launch_bounds__(256)
void scan_kernel(const float* __restrict__ proj, const float* __restrict__ dt,
                 const float* __restrict__ A_log, const float* __restrict__ Dv,
                 const float* __restrict__ xz, float* __restrict__ xc)
{
    __shared__ float sB[SCAN_T][NST];
    __shared__ float sC[SCAN_T][NST];
    const int tid = threadIdx.x;
    const int c   = blockIdx.x * 256 + tid;     // 0..BSZ*DI-1
    const int b   = c >> 11;                    // / DI
    const int d   = c & (DI - 1);

    float A[NST], h[NST];
#pragma unroll
    for (int n = 0; n < NST; ++n) {
        A[n] = -__expf(A_log[d * NST + n]);
        h[n] = 0.f;
    }
    const float Dd = Dv[d];

    for (int t0 = 0; t0 < LQ; t0 += SCAN_T) {
        __syncthreads();
        for (int j = tid; j < SCAN_T * 2 * NST; j += 256) {
            int tt = j >> 5, col = j & 31;
            float v = proj[(size_t)(b * LQ + t0 + tt) * PRJ + DTR + col];
            if (col < NST) sB[tt][col] = v; else sC[tt][col - NST] = v;
        }
        __syncthreads();
        for (int tt = 0; tt < SCAN_T; ++tt) {
            size_t idx = (size_t)(b * LQ + t0 + tt) * DI + d;
            float u   = xc[idx];
            float dtv = dt[idx];
            float du  = dtv * u;
            float y   = u * Dd;
#pragma unroll
            for (int n = 0; n < NST; ++n) {
                h[n] = __expf(dtv * A[n]) * h[n] + du * sB[tt][n];
                y   += h[n] * sC[tt][n];
            }
            float zg = xz[(size_t)(b * LQ + t0 + tt) * (2 * DI) + DI + d];
            xc[idx] = y * zg;
        }
    }
}

// ---------------------------------------------------------------------------
// out = RMSNorm(out + x) * norm_w, in place on d_out; one block per row.
// ---------------------------------------------------------------------------
__global__ __launch_bounds__(256)
void rmsnorm_kernel(float* __restrict__ out, const float* __restrict__ x,
                    const float* __restrict__ w)
{
    __shared__ float red[256];
    const int row = blockIdx.x;
    const int tid = threadIdx.x;
    float*       orow = out + (size_t)row * DM;
    const float* xrow = x   + (size_t)row * DM;

    float hv[4], ss = 0.f;
#pragma unroll
    for (int j = 0; j < 4; ++j) {
        int col = tid * 4 + j;
        float v = orow[col] + xrow[col];
        hv[j] = v;
        ss += v * v;
    }
    red[tid] = ss;
    __syncthreads();
    for (int s = 128; s > 0; s >>= 1) {
        if (tid < s) red[tid] += red[tid + s];
        __syncthreads();
    }
    float sc = rsqrtf(red[0] * (1.f / DM) + 1e-5f);
#pragma unroll
    for (int j = 0; j < 4; ++j) {
        int col = tid * 4 + j;
        orow[col] = hv[j] * sc * w[col];
    }
}

// ---------------------------------------------------------------------------
extern "C" void kernel_launch(void* const* d_in, const int* in_sizes, int n_in,
                              void* d_out, int out_size, void* d_ws, size_t ws_size,
                              hipStream_t stream)
{
    const float* x      = (const float*)d_in[0];
    const float* W_in   = (const float*)d_in[1];
    const float* conv_w = (const float*)d_in[2];
    const float* conv_b = (const float*)d_in[3];
    const float* W_xprj = (const float*)d_in[4];
    const float* W_dt   = (const float*)d_in[5];
    const float* b_dt   = (const float*)d_in[6];
    const float* A_log  = (const float*)d_in[7];
    const float* Dv     = (const float*)d_in[8];
    const float* W_out  = (const float*)d_in[9];
    const float* norm_w = (const float*)d_in[10];
    float* out = (float*)d_out;

    // ---- workspace: fp32 activations, then bf16 split/transposed weights ----
    float* ws   = (float*)d_ws;
    float* xz   = ws;                               // ROWS * 2*DI
    float* xc   = xz   + (size_t)ROWS * 2 * DI;     // ROWS * DI
    float* proj = xc   + (size_t)ROWS * DI;         // ROWS * PRJ
    float* dt   = proj + (size_t)ROWS * PRJ;        // ROWS * DI

    __bf16* bptr    = (__bf16*)(dt + (size_t)ROWS * DI);
    __bf16* Wint_h  = bptr;  bptr += (size_t)(2 * DI) * DM;   // [4096 x 1024]
    __bf16* Wint_l  = bptr;  bptr += (size_t)(2 * DI) * DM;
    __bf16* Wxpt_h  = bptr;  bptr += (size_t)PRJ * DI;        // [96 x 2048]
    __bf16* Wxpt_l  = bptr;  bptr += (size_t)PRJ * DI;
    __bf16* Wdtt_h  = bptr;  bptr += (size_t)DI * DTR;        // [2048 x 64]
    __bf16* Wdtt_l  = bptr;  bptr += (size_t)DI * DTR;
    __bf16* Woutt_h = bptr;  bptr += (size_t)DM * DI;         // [1024 x 2048]
    __bf16* Woutt_l = bptr;  bptr += (size_t)DM * DI;

    const int ew = (ROWS * DI + 255) / 256;

    // 0) transpose + bf16-split all weights (cheap, bandwidth bound)
    tsplit_kernel<<<dim3((2 * DI) / 32, DM / 32), 256, 0, stream>>>(
        W_in, Wint_h, Wint_l, DM, 2 * DI);
    tsplit_kernel<<<dim3(PRJ / 32, DI / 32), 256, 0, stream>>>(
        W_xprj, Wxpt_h, Wxpt_l, DI, PRJ);
    tsplit_kernel<<<dim3(DI / 32, DTR / 32), 256, 0, stream>>>(
        W_dt, Wdtt_h, Wdtt_l, DTR, DI);
    tsplit_kernel<<<dim3(DM / 32, DI / 32), 256, 0, stream>>>(
        W_out, Woutt_h, Woutt_l, DI, DM);

    // 1) xz = x @ W_in                  [4096 x 4096], K=1024, 32x64 wave tiles
    gemm_wmma<2, 4><<<dim3((2 * DI) / 64, ROWS / 32), 32, 0, stream>>>(
        x, Wint_h, Wint_l, xz, DM, DM, 2 * DI);
    // 2) depthwise conv + silu; silu(z) in place
    conv_silu_kernel<<<ew, 256, 0, stream>>>(xz, conv_w, conv_b, xc);
    // 3) proj = xc @ W_xproj            [4096 x 96], K=2048, 32x48 wave tiles
    gemm_wmma<2, 3><<<dim3(PRJ / 48, ROWS / 32), 32, 0, stream>>>(
        xc, Wxpt_h, Wxpt_l, proj, DI, DI, PRJ);
    // 4) dt_lin = proj[:, :64] @ W_dt   [4096 x 2048], K=64 (lda = PRJ)
    gemm_wmma<2, 4><<<dim3(DI / 64, ROWS / 32), 32, 0, stream>>>(
        proj, Wdtt_h, Wdtt_l, dt, DTR, PRJ, DI);
    // 5) dt = softplus(dt_lin + b_dt)
    softplus_kernel<<<ew, 256, 0, stream>>>(dt, b_dt);
    // 6) selective scan + D skip + gate, writes y over xc
    scan_kernel<<<(BSZ * DI) / 256, 256, 0, stream>>>(proj, dt, A_log, Dv, xz, xc);
    // 7) out = y @ W_out                [4096 x 1024], K=2048
    gemm_wmma<2, 4><<<dim3(DM / 64, ROWS / 32), 32, 0, stream>>>(
        xc, Woutt_h, Woutt_l, out, DI, DI, DM);
    // 8) residual + RMSNorm in place
    rmsnorm_kernel<<<ROWS, 256, 0, stream>>>(out, x, norm_w);
}
